// loss_func_43112881717767
// MI455X (gfx1250) — matrix operand order repaired and checked by reading, not compiled, not run on hardware
//
#include <hip/hip_runtime.h>
#include <hip/hip_bf16.h>

// ---------------------------------------------------------------------------
// loss = mean((output-label)^2) + 0.01*sum(adj) + (tr(expm(adj*adj)) - n)
// tr(expm(A)) - n = sum_{k>=1} tr(A^k)/k!  with B_k = A^k/k!,
// B_{k+1} = (A*B_k)/(k+1), via bf16x3-split WMMA GEMMs (fp32-class accuracy
// on the high-throughput v_wmma_f32_16x16x32_bf16 pipe).
// Each wave computes a 16x32 output tile (2 accumulators) so the A fragments
// are reused and the two independent WMMA chains hide the XDL RAW latency.
// ---------------------------------------------------------------------------

typedef __attribute__((ext_vector_type(4)))  float   v4f;
typedef __attribute__((ext_vector_type(8)))  float   v8f;
typedef __attribute__((ext_vector_type(8)))  __bf16  v8bf;
typedef __attribute__((ext_vector_type(16))) __bf16  v16bf;

#define N_MAT   1024
#define MSE_N   (16384 * 1024)
#define K_TERMS 30          // Taylor terms; ||A||~2.6 -> remainder < 1e-12

// ---------------- accumulator init ----------------
__global__ void init_kernel(float* acc) {
    if (threadIdx.x < 8) acc[threadIdx.x] = 0.0f;
}

// ---------------- L_re: MSE reduction (bandwidth-bound, NT loads) ----------
__global__ void __launch_bounds__(256)
mse_kernel(const float* __restrict__ o, const float* __restrict__ l,
           float* __restrict__ acc) {
    __shared__ float red[256];
    const v4f* o4 = (const v4f*)o;
    const v4f* l4 = (const v4f*)l;
    const int n4 = MSE_N / 4;
    float s = 0.0f;
    for (int i = blockIdx.x * blockDim.x + threadIdx.x; i < n4;
         i += gridDim.x * blockDim.x) {
        v4f a = __builtin_nontemporal_load(&o4[i]);   // TH=NT: don't evict the
        v4f b = __builtin_nontemporal_load(&l4[i]);   // L2-resident GEMM set
        v4f d = a - b;
        s += d.x * d.x + d.y * d.y + d.z * d.z + d.w * d.w;
    }
    red[threadIdx.x] = s;
    __syncthreads();
    for (int off = 128; off > 0; off >>= 1) {
        if (threadIdx.x < off) red[threadIdx.x] += red[threadIdx.x + off];
        __syncthreads();
    }
    if (threadIdx.x == 0) atomicAdd(&acc[0], red[0]);
}

// ---------------- prep: L_sp sum, A = adj*adj -> bf16 hi/lo, B1 = A (cm),
// ---------------- and tr(B1) seed ----------------
__global__ void __launch_bounds__(256)
prep_kernel(const float* __restrict__ adj,
            __bf16* __restrict__ Ahi, __bf16* __restrict__ Alo,
            __bf16* __restrict__ Bhi, __bf16* __restrict__ Blo,
            float* __restrict__ acc) {
    __shared__ float red[256];
    const int i = blockIdx.x * blockDim.x + threadIdx.x;   // N_MAT*N_MAT threads
    const float a = adj[i];
    const float A = a * a;
    const __bf16 h  = (__bf16)A;
    const __bf16 lo = (__bf16)(A - (float)h);
    const int row = i >> 10;
    const int col = i & (N_MAT - 1);
    Ahi[i] = h;                          // row-major A (fixed left operand)
    Alo[i] = lo;
    const int cm = col * N_MAT + row;    // column-major B_1
    Bhi[cm] = h;
    Blo[cm] = lo;
    if (row == col) atomicAdd(&acc[2], A);   // tr(B_1) = tr(A)
    red[threadIdx.x] = a;
    __syncthreads();
    for (int off = 128; off > 0; off >>= 1) {
        if (threadIdx.x < off) red[threadIdx.x] += red[threadIdx.x + off];
        __syncthreads();
    }
    if (threadIdx.x == 0) atomicAdd(&acc[1], red[0]);      // L_sp partial
}

// ---------------- WMMA fragment loads ----------------
// A (row-major, contiguous k order used here, same as B): lane half 0 holds K=[0..15],
// half 1 holds K=[16..31]; 2x b128 loads per lane.
__device__ __forceinline__ v16bf load_fragA(const __bf16* __restrict__ rowp,
                                            int k0, int half) {
    v8bf a = *(const v8bf*)(rowp + k0 + 16 * half);
    v8bf b = *(const v8bf*)(rowp + k0 + 16 * half + 8);
    return __builtin_shufflevector(a, b, 0, 1, 2, 3, 4, 5, 6, 7,
                                   8, 9, 10, 11, 12, 13, 14, 15);
}

// B (column-major storage, contiguous k order used here, same as A): lanes 0-15 hold K=[0..15],
// lanes 16-31 hold K=[16..31] of column N=lane&15; contiguous 32B per lane.
__device__ __forceinline__ v16bf load_fragB(const __bf16* __restrict__ colp,
                                            int k0, int half) {
    const v8bf* p = (const v8bf*)(colp + k0 + 16 * half);
    v8bf a = p[0];
    v8bf b = p[1];
    return __builtin_shufflevector(a, b, 0, 1, 2, 3, 4, 5, 6, 7,
                                   8, 9, 10, 11, 12, 13, 14, 15);
}

// Epilogue for one 16x16 tile: scale, bf16 hi/lo split for next iteration,
// column-major b128 stores, trace atomics for diagonal tiles.
__device__ __forceinline__ void tile_epilogue(v8f accv, float invk, int c,
                                              int m0, int half, int lane,
                                              __bf16* __restrict__ Chi,
                                              __bf16* __restrict__ Clo,
                                              float* __restrict__ acc,
                                              bool diagTile) {
    float vals[8];
    v8bf hi, lo;
#pragma unroll
    for (int v = 0; v < 8; ++v) {
        float x = accv[v] * invk;
        vals[v] = x;
        __bf16 h = (__bf16)x;
        hi[v] = h;
        lo[v] = (__bf16)(x - (float)h);
    }
    // C/D layout: VGPR v, lane -> (M = m0 + 8*half + v, N = c); column-major
    // store => 8 consecutive elements per lane.
    __bf16* ch = Chi + (size_t)c * N_MAT + m0 + 8 * half;
    __bf16* cl = Clo + (size_t)c * N_MAT + m0 + 8 * half;
    *(v8bf*)ch = hi;
    *(v8bf*)cl = lo;
    // Diagonal elements live in lanes 0-7 (v=lane) and 24-31 (v=lane-24).
    if (diagTile && ((lane >> 4) == ((lane >> 3) & 1)))
        atomicAdd(&acc[2], vals[lane & 7]);
}

// ---------------- C = (A*B) * invk: 16x32 tile per wave, bf16x3 split -------
__global__ void __launch_bounds__(256)
gemm_bf16x3_kernel(const __bf16* __restrict__ Ahi, const __bf16* __restrict__ Alo,
                   const __bf16* __restrict__ Bhi, const __bf16* __restrict__ Blo,
                   __bf16* __restrict__ Chi, __bf16* __restrict__ Clo,
                   float* __restrict__ acc, float invk) {
    const int lane = threadIdx.x & 31;
    const int wave = (blockIdx.x * blockDim.x + threadIdx.x) >> 5;  // 0..2047
    const int m0 = (wave & 63) << 4;        // 64 M-tiles
    const int n0 = (wave >> 6) << 5;        // 32 N-supertiles of 32 columns
    const int half = lane >> 4;
    const int r  = m0 + (lane & 15);
    const int c0 = n0 + (lane & 15);
    const int c1 = c0 + 16;

    const __bf16* arh  = Ahi + (size_t)r  * N_MAT;
    const __bf16* arl  = Alo + (size_t)r  * N_MAT;
    const __bf16* b0hp = Bhi + (size_t)c0 * N_MAT;   // column c0 contiguous
    const __bf16* b0lp = Blo + (size_t)c0 * N_MAT;
    const __bf16* b1hp = Bhi + (size_t)c1 * N_MAT;   // column c1 contiguous
    const __bf16* b1lp = Blo + (size_t)c1 * N_MAT;

    v8f acc0 = {};
    v8f acc1 = {};
    for (int k0 = 0; k0 < N_MAT; k0 += 32) {
        v16bf ah  = load_fragA(arh,  k0, half);
        v16bf al  = load_fragA(arl,  k0, half);
        v16bf b0h = load_fragB(b0hp, k0, half);
        v16bf b0l = load_fragB(b0lp, k0, half);
        v16bf b1h = load_fragB(b1hp, k0, half);
        v16bf b1l = load_fragB(b1lp, k0, half);
        // Two independent accumulation chains keep the XDL pipe busy.
        acc0 = __builtin_amdgcn_wmma_f32_16x16x32_bf16(false, al, false, b0h,
                                                       (short)0, acc0, false, false);
        acc1 = __builtin_amdgcn_wmma_f32_16x16x32_bf16(false, al, false, b1h,
                                                       (short)0, acc1, false, false);
        acc0 = __builtin_amdgcn_wmma_f32_16x16x32_bf16(false, ah, false, b0l,
                                                       (short)0, acc0, false, false);
        acc1 = __builtin_amdgcn_wmma_f32_16x16x32_bf16(false, ah, false, b1l,
                                                       (short)0, acc1, false, false);
        acc0 = __builtin_amdgcn_wmma_f32_16x16x32_bf16(false, ah, false, b0h,
                                                       (short)0, acc0, false, false);
        acc1 = __builtin_amdgcn_wmma_f32_16x16x32_bf16(false, ah, false, b1h,
                                                       (short)0, acc1, false, false);
    }

    tile_epilogue(acc0, invk, c0, m0, half, lane, Chi, Clo, acc, m0 == n0);
    tile_epilogue(acc1, invk, c1, m0, half, lane, Chi, Clo, acc, m0 == n0 + 16);
}

// ---------------- combine ----------------
__global__ void finalize_kernel(const float* __restrict__ acc,
                                float* __restrict__ out) {
    // L_re (mean) + 0.01 * L_sp + 1.0 * L_dag
    out[0] = acc[0] * (1.0f / (float)MSE_N) + 0.01f * acc[1] + acc[2];
}

extern "C" void kernel_launch(void* const* d_in, const int* in_sizes, int n_in,
                              void* d_out, int out_size, void* d_ws, size_t ws_size,
                              hipStream_t stream) {
    const float* output = (const float*)d_in[0];
    const float* adj    = (const float*)d_in[1];
    const float* label  = (const float*)d_in[2];
    // d_in[3] soft_threshold: unused, matching the reference.
    float* out = (float*)d_out;

    char* ws = (char*)d_ws;
    float* acc = (float*)ws;                      // [0]=mse [1]=sp [2]=trace
    const size_t M = (size_t)N_MAT * N_MAT;       // 1M elements
    __bf16* Ahi = (__bf16*)(ws + 256);
    __bf16* Alo = Ahi + M;
    __bf16* B0h = Alo + M;
    __bf16* B0l = B0h + M;
    __bf16* B1h = B0l + M;
    __bf16* B1l = B1h + M;

    init_kernel<<<1, 32, 0, stream>>>(acc);
    mse_kernel<<<2048, 256, 0, stream>>>(output, label, acc);
    prep_kernel<<<(int)(M / 256), 256, 0, stream>>>(adj, Ahi, Alo, B0h, B0l, acc);

    __bf16 *sh = B0h, *sl = B0l, *dh = B1h, *dl = B1l;
    for (int k = 2; k <= K_TERMS; ++k) {
        // 2048 waves (16x32 tile each) = 256 blocks x 8 wave32s
        gemm_bf16x3_kernel<<<256, 256, 0, stream>>>(Ahi, Alo, sh, sl, dh, dl,
                                                    acc, 1.0f / (float)k);
        __bf16* t;
        t = sh; sh = dh; dh = t;
        t = sl; sl = dl; dl = t;
    }
    finalize_kernel<<<1, 1, 0, stream>>>(acc, out);
}
